// GraphSelfAttentionLayer_90434831385335
// MI455X (gfx1250) — matrix operand order repaired and checked by reading, not compiled, not run on hardware
//
#include <hip/hip_runtime.h>

// GraphSelfAttentionLayer, MI455X/gfx1250, wave32 + v_wmma_f32_16x16x32_bf16.
// Folded formulation:
//   WqKT[b][c=h*32+k][f] = sum_d Wq[f,h*96+d] * k[b,k,h*96+d]      (bf16, L2-resident)
//   aff_raw[b,n,c]       = roi[b,n,:] @ WqKT[b]                    (bf16 WMMA, dominant)
//   vWT[b][c=h*96+g][k]  = sum_f v[b,k,f] * Wout[h,g,f]            (bf16, L2-resident)
//   out[b,n,h*96+g]      = softmax_k(...) @ vWT + bout             (bf16 WMMA)
// Workspace need: ~10.24 MB.

#define B_     16
#define N_     1024
#define FEAT_  768
#define H_     8
#define NONGT_ 20
#define M_     10
#define POS_   64
#define DG_    96
#define K30    30
#define KP     32
#define CTOT   256            // H_*KP
#define NEGV   (-9.0e15f)

#define APITCH 776            // bf16 elems, 1552B row stride: 16B aligned, bank-conflict free
#define PPITCH 40             // bf16 elems, 80B row stride

typedef __attribute__((ext_vector_type(16))) __bf16 v16bf;
typedef __attribute__((ext_vector_type(8)))  __bf16 v8bf;
typedef __attribute__((ext_vector_type(8)))  float  v8f;

__device__ __forceinline__ __bf16 f2bf(float f) {
  union { float f; unsigned u; } x; x.f = f;
  unsigned r = x.u + 0x7FFFu + ((x.u >> 16) & 1u);   // round-to-nearest-even
  unsigned short hbits = (unsigned short)(r >> 16);
  __bf16 o; __builtin_memcpy(&o, &hbits, 2);
  return o;
}

// ---------------------------------------------------------------------------
// P1: per (b, kr<32) row: v row, k row (nongt GEMM / memory slots), qb term.
// Rows 30,31 are zero padding so downstream GEMMs need no masking.
// ---------------------------------------------------------------------------
__global__ __launch_bounds__(256) void gsa_prep_rows(
    const float* __restrict__ roi, const float* __restrict__ aux,
    const float* __restrict__ Wk,  const float* __restrict__ bk,
    const float* __restrict__ mk,  const float* __restrict__ mv,
    const float* __restrict__ bq,
    float* __restrict__ kmat, float* __restrict__ vmat, float* __restrict__ qb)
{
  __shared__ float vrow[FEAT_];
  __shared__ float krow[FEAT_];
  const int kr = blockIdx.x;     // 0..31
  const int b  = blockIdx.y;
  const int t  = threadIdx.x;
  const float scm = sqrtf((float)FEAT_ / (float)H_);   // sqrt(96)

  for (int i = t; i < FEAT_; i += 256) {
    float v;
    if      (kr < NONGT_) v = roi[((long)b*N_ + kr)*FEAT_ + i];
    else if (kr < K30)    v = scm * mv[(kr-NONGT_)*FEAT_ + i] * aux[b*FEAT_ + i];
    else                  v = 0.0f;
    vrow[i] = v;
  }
  __syncthreads();
  for (int i = t; i < FEAT_; i += 256) vmat[((long)b*KP + kr)*FEAT_ + i] = vrow[i];

  if (kr < NONGT_) {
    for (int c = t; c < FEAT_; c += 256) {       // coalesced over Wk columns
      float s = bk[c];
      for (int f = 0; f < FEAT_; ++f) s = fmaf(vrow[f], Wk[(long)f*FEAT_ + c], s);
      krow[c] = s;
    }
  } else if (kr < K30) {
    for (int c = t; c < FEAT_; c += 256)
      krow[c] = scm * mk[(kr-NONGT_)*FEAT_ + c] * aux[b*FEAT_ + c];
  } else {
    for (int c = t; c < FEAT_; c += 256) krow[c] = 0.0f;
  }
  __syncthreads();
  for (int c = t; c < FEAT_; c += 256) kmat[((long)b*KP + kr)*FEAT_ + c] = krow[c];
  if (t < H_) {
    float s = 0.0f;
    for (int d = 0; d < DG_; ++d) s = fmaf(bq[t*DG_ + d], krow[t*DG_ + d], s);
    qb[(long)b*CTOT + t*KP + kr] = s;
  }
}

// ---------------------------------------------------------------------------
// P2: vWT[b][c=h*96+g][k<32] = v[b,k,:] . Wout[c,:]   (bf16 out, zero rows 30,31)
// ---------------------------------------------------------------------------
__global__ __launch_bounds__(256) void gsa_prep_vwt(
    const float* __restrict__ vmat, const float* __restrict__ Wout,
    __bf16* __restrict__ vwt)
{
  __shared__ float wrow[FEAT_];
  const int c = blockIdx.x;     // 0..767
  const int b = blockIdx.y;
  const int t = threadIdx.x;
  for (int i = t; i < FEAT_; i += 256) wrow[i] = Wout[(long)c*FEAT_ + i];
  __syncthreads();
  const int k = t >> 3, part = t & 7;            // 32 k * 8 partial lanes
  const float* vr = vmat + ((long)b*KP + k)*FEAT_ + part*96;
  const float* wr = wrow + part*96;
  float s = 0.0f;
  for (int i = 0; i < 96; ++i) s = fmaf(vr[i], wr[i], s);
  s += __shfl_xor(s, 1, 32);
  s += __shfl_xor(s, 2, 32);
  s += __shfl_xor(s, 4, 32);
  if (part == 0) vwt[((long)b*FEAT_ + c)*KP + k] = f2bf(s);
}

// ---------------------------------------------------------------------------
// P3: WqKT[b][c=h*32+k][f] = sum_d Wq[f,h*96+d]*kmat[b,k,h*96+d]  (bf16 out)
// ---------------------------------------------------------------------------
__global__ __launch_bounds__(256) void gsa_prep_wqkt(
    const float* __restrict__ kmat, const float* __restrict__ Wq,
    __bf16* __restrict__ wqkt)
{
  __shared__ float kls[K30*97];
  __shared__ float wqt[96*97];
  const int h = blockIdx.x, b = blockIdx.y, t = threadIdx.x;
  for (int i = t; i < K30*DG_; i += 256) {
    int kk = i / DG_, d = i - kk*DG_;
    kls[kk*97 + d] = kmat[((long)b*KP + kk)*FEAT_ + h*DG_ + d];
  }
  const int k = t & 31, fl = t >> 5;             // fl 0..7
  for (int ft = 0; ft < 8; ++ft) {
    const int f0 = ft*96;
    __syncthreads();
    for (int i = t; i < 96*DG_; i += 256) {      // stage Wq tile, coalesced
      int fr = i / DG_, d = i - fr*DG_;
      wqt[fr*97 + d] = Wq[(long)(f0+fr)*FEAT_ + h*DG_ + d];
    }
    __syncthreads();
    for (int i = 0; i < 12; ++i) {
      const int fr = fl + 8*i;
      float s = 0.0f;
      if (k < K30) {
        for (int d = 0; d < DG_; ++d) s = fmaf(wqt[fr*97 + d], kls[k*97 + d], s);
      }
      wqkt[((long)b*CTOT + h*KP + k)*FEAT_ + f0 + fr] = f2bf(s);
    }
  }
}

// ---------------------------------------------------------------------------
// Main fused attention: block = (b, 16-row n tile), 8 waves, wave w == head h.
// ---------------------------------------------------------------------------
__global__ __launch_bounds__(256) void gsa_attn_main(
    const float* __restrict__ roi,  const int* __restrict__ adj,
    const float* __restrict__ pe,   const float* __restrict__ lbl,
    const float* __restrict__ Wpos, const float* __restrict__ bpos,
    const float* __restrict__ bout, const float* __restrict__ qb,
    const __bf16* __restrict__ wqkt, const __bf16* __restrict__ vwt,
    float* __restrict__ out)
{
  __shared__ __align__(16) __bf16 aLds[16*APITCH];
  __shared__ __align__(16) __bf16 pLds[8*16*PPITCH];
  __shared__ float bias2[16*H_*NONGT_];
  __shared__ float addt [16*NONGT_];
  __shared__ float wposL[POS_*H_];
  __shared__ float bposL[H_];

  const int t  = threadIdx.x;
  const int b  = blockIdx.y;
  const int n0 = blockIdx.x * 16;

  // ---- stage & convert A tile (16 x 768 f32 -> bf16, WMMA A layout source) ----
  const float* rbase = roi + ((long)b*N_ + n0)*FEAT_;
  for (int i = t; i < 16*FEAT_; i += 256) {
    int r = i / FEAT_, c = i - r*FEAT_;
    aLds[r*APITCH + c] = f2bf(rbase[i]);
  }
  for (int i = t; i < POS_*H_; i += 256) wposL[i] = Wpos[i];
  if (t < H_) bposL[t] = bpos[t];
  for (int i = t; i < 16*NONGT_; i += 256) {
    long a = ((long)b*N_ + n0 + i/NONGT_)*NONGT_ + (i % NONGT_);
    addt[i] = (adj[a] > 0) ? lbl[a] : NEGV;     // masked slots absorb to NEG
  }
  __syncthreads();

  // ---- positional bias: bias2[n][h][k] = log(max(relu(pe.Wpos+bpos),1e-6)) ----
  for (int p = t; p < 16*NONGT_; p += 256) {
    int n = p / NONGT_, kk = p - n*NONGT_;
    const float4* p4 = (const float4*)(pe + (((long)b*N_ + n0 + n)*NONGT_ + kk)*POS_);
    float s[H_];
    for (int h = 0; h < H_; ++h) s[h] = bposL[h];
    for (int q = 0; q < POS_/4; ++q) {
      float4 v = p4[q];
      const float* w0 = wposL + (q*4)*H_;
      for (int h = 0; h < H_; ++h)
        s[h] += v.x*w0[h] + v.y*w0[H_+h] + v.z*w0[2*H_+h] + v.w*w0[3*H_+h];
    }
    for (int h = 0; h < H_; ++h)
      bias2[(n*H_ + h)*NONGT_ + kk] = logf(fmaxf(s[h], 1e-6f));
  }
  __syncthreads();

  // ---- per-wave: aff = roi_tile @ WqKT[b][head cols], K=768 in 24 bf16 WMMA steps
  const int w    = t >> 5;          // wave index == head
  const int lane = t & 31;
  const int rowA = lane & 15;
  const int aoff = (lane < 16) ? 0 : 16;   // A layout K split per lane-half
  const int boff = (lane < 16) ? 0 : 16;   // B layout K split per lane-half

  v8f acc0 = {}, acc1 = {};
  const __bf16* bp0  = wqkt + ((long)b*CTOT + w*KP + rowA)*FEAT_ + boff;
  const __bf16* bp1  = bp0 + 16*FEAT_;
  const __bf16* arow = aLds + rowA*APITCH + aoff;
  for (int ks = 0; ks < FEAT_/32; ++ks) {
    const int kb = ks*32;
    v8bf alo = *(const v8bf*)(arow + kb);
    v8bf ahi = *(const v8bf*)(arow + kb + 8);
    v16bf av;
    #pragma unroll
    for (int e = 0; e < 8; ++e) { av[e] = alo[e]; av[e+8] = ahi[e]; }
    v16bf b0 = *(const v16bf*)(bp0 + kb);
    v16bf b1 = *(const v16bf*)(bp1 + kb);
    if (ks + 1 < FEAT_/32) {
      __builtin_prefetch(bp0 + kb + 32, 0, 0);   // global_prefetch next B step
      __builtin_prefetch(bp1 + kb + 32, 0, 0);
    }
    acc0 = __builtin_amdgcn_wmma_f32_16x16x32_bf16(false, av, false, b0, (short)0, acc0, false, false);
    acc1 = __builtin_amdgcn_wmma_f32_16x16x32_bf16(false, av, false, b1, (short)0, acc1, false, false);
  }

  // ---- epilogue: scale + qb + pos-bias + mask + label, softmax over 30 slots ----
  const int   kcol  = lane & 15;              // acc0 -> k=kcol, acc1 -> k=16+kcol
  const int   rhalf = (lane < 16) ? 0 : 8;
  const float scale = rsqrtf((float)DG_);
  const float qv0 = qb[(long)b*CTOT + w*KP + kcol];
  const float qv1 = qb[(long)b*CTOT + w*KP + 16 + kcol];
  const int   k2  = 16 + kcol;
  float p0v[8], p1v[8];
  #pragma unroll
  for (int j = 0; j < 8; ++j) {
    const int n = j + rhalf;
    float a0 = (acc0[j] + qv0) * scale;
    float a1 = (acc1[j] + qv1) * scale;
    float w0 = a0 + bias2[(n*H_ + w)*NONGT_ + kcol] + addt[n*NONGT_ + kcol];
    float w1;
    if      (k2 < NONGT_) w1 = a1 + bias2[(n*H_ + w)*NONGT_ + k2] + addt[n*NONGT_ + k2];
    else if (k2 < K30)    w1 = a1;          // memory slots: raw aff
    else                  w1 = NEGV;        // pad cols -> p = 0
    float m = fmaxf(w0, w1);
    for (int off = 1; off < 16; off <<= 1) m = fmaxf(m, __shfl_xor(m, off, 32));
    float e0 = __expf(w0 - m);
    float e1 = __expf(w1 - m);
    float s = e0 + e1;
    for (int off = 1; off < 16; off <<= 1) s += __shfl_xor(s, off, 32);
    float inv = 1.0f / s;
    p0v[j] = e0 * inv;
    p1v[j] = e1 * inv;
  }

  // ---- p (C layout) -> LDS bf16 -> reload in A layout ----
  __bf16* pw = pLds + w*16*PPITCH;
  #pragma unroll
  for (int j = 0; j < 8; ++j) {
    const int n = j + rhalf;
    pw[n*PPITCH + kcol]      = f2bf(p0v[j]);
    pw[n*PPITCH + 16 + kcol] = f2bf(p1v[j]);
  }
  __syncthreads();
  v8bf plo = *(const v8bf*)(pw + rowA*PPITCH + aoff);
  v8bf phi = *(const v8bf*)(pw + rowA*PPITCH + aoff + 8);
  v16bf pa;
  #pragma unroll
  for (int e = 0; e < 8; ++e) { pa[e] = plo[e]; pa[e+8] = phi[e]; }

  // ---- out = p @ vW + bout : one K=32 bf16 WMMA per 16-col tile ----
  const v8f zero = {};
  for (int ct = 0; ct < 6; ++ct) {
    const int gcol = ct*16 + kcol;
    const __bf16* bvp = vwt + ((long)b*FEAT_ + w*DG_ + gcol)*KP + boff;
    v16bf bv = *(const v16bf*)bvp;
    v8f d = __builtin_amdgcn_wmma_f32_16x16x32_bf16(false, pa, false, bv, (short)0, zero, false, false);
    const float bo = bout[w*DG_ + gcol];
    #pragma unroll
    for (int j = 0; j < 8; ++j) {
      const int n = j + rhalf;
      out[((long)b*N_ + n0 + n)*FEAT_ + w*DG_ + gcol] = d[j] + bo;
    }
  }
}

// ---------------------------------------------------------------------------
extern "C" void kernel_launch(void* const* d_in, const int* in_sizes, int n_in,
                              void* d_out, int out_size, void* d_ws, size_t ws_size,
                              hipStream_t stream)
{
  (void)in_sizes; (void)n_in; (void)out_size; (void)ws_size;
  const float* roi  = (const float*)d_in[0];
  const int*   adj  = (const int*)  d_in[1];
  const float* pe   = (const float*)d_in[2];
  const float* lbl  = (const float*)d_in[3];
  const float* aux  = (const float*)d_in[4];
  const float* Wq   = (const float*)d_in[5];
  const float* bq   = (const float*)d_in[6];
  const float* Wk   = (const float*)d_in[7];
  const float* bk   = (const float*)d_in[8];
  const float* Wpos = (const float*)d_in[9];
  const float* bpos = (const float*)d_in[10];
  const float* mk   = (const float*)d_in[11];
  const float* mv   = (const float*)d_in[12];
  const float* Wout = (const float*)d_in[13];
  const float* bout = (const float*)d_in[14];
  float* out = (float*)d_out;

  char* ws = (char*)d_ws;                         // ~10.24 MB used, 256B aligned slabs
  float*  kmat = (float*) (ws + 0);               // 16*32*768*4 = 1,572,864
  float*  vmat = (float*) (ws + 1572864);         // 1,572,864
  float*  qbuf = (float*) (ws + 3145728);         // 16*256*4 = 16,384
  __bf16* vwt  = (__bf16*)(ws + 3162112);         // 16*768*32*2 = 786,432
  __bf16* wqkt = (__bf16*)(ws + 3948544);         // 16*256*768*2 = 6,291,456

  gsa_prep_rows<<<dim3(32, B_),  256, 0, stream>>>(roi, aux, Wk, bk, mk, mv, bq,
                                                   kmat, vmat, qbuf);
  gsa_prep_vwt <<<dim3(768, B_), 256, 0, stream>>>(vmat, Wout, vwt);
  gsa_prep_wqkt<<<dim3(8, B_),   256, 0, stream>>>(kmat, Wq, wqkt);
  gsa_attn_main<<<dim3(N_/16, B_), 256, 0, stream>>>(roi, adj, pe, lbl, Wpos, bpos,
                                                     bout, qbuf, wqkt, vwt, out);
}
